// FPT_UpsampleBlock_33079838114109
// MI455X (gfx1250) — matrix operand order repaired and checked by reading, not compiled, not run on hardware
//
#include <hip/hip_runtime.h>
#include <hip/hip_bf16.h>

// ---------------------------------------------------------------------------
// FPT Upsample block for MI455X (gfx1250, wave32)
//   - all channel-mixing ops as f16 WMMA GEMMs (v_wmma_f32_16x16x32_f16)
//   - activations kept (points x channels) row-major so LDS staging is a
//     straight block copy done with GLOBAL_LOAD_ASYNC_TO_LDS_B128 (ASYNCcnt),
//     double-buffered against the WMMA consumption
//   - B fragments batched before the WMMAs so the matrix ops issue
//     back-to-back behind a single s_wait_dscnt
//   - flash attention per (batch, head) with WMMA QK^T / PV + online softmax
// ---------------------------------------------------------------------------

typedef _Float16 h16;
typedef __attribute__((ext_vector_type(16))) _Float16 v16h;
typedef __attribute__((ext_vector_type(8)))  float    v8f;

#define BATCH  8
#define NPTS   2048
#define MPTS   512
#define BN     (BATCH * NPTS)   // 16384
#define CIN    256
#define CFP    128
#define COUT   128
#define CG     512
#define KFU    544              // 3 + 512 padded to multiple of 32
#define NHEADS 8
#define DH     16

#define FLAG_RELU  1
#define FLAG_F16   2
#define FLAG_BATCH 4

// ------------------------- async copy helpers (CDNA5) ----------------------
// 64 rows x 32 halves tile, global row stride = strideHalves.
// 256 chunks of 16B; each lane DMAs its chunks straight into LDS (no VGPR
// round trip, tracked by ASYNCcnt).
__device__ inline void async_copy_tile64x32(const h16* __restrict__ gsrc,
                                            int strideHalves, h16* ldsDst)
{
#pragma unroll
    for (int t = 0; t < 2; ++t) {
        int s = threadIdx.x + t * 128;          // 0..255
        int i = s >> 2, seg = s & 3;
        const h16* g = gsrc + (size_t)i * strideHalves + seg * 8;
        unsigned lds = (unsigned)(uintptr_t)(ldsDst + i * 32 + seg * 8);
        unsigned long long ga = (unsigned long long)(uintptr_t)g;
        asm volatile("global_load_async_to_lds_b128 %0, %1, off"
                     :: "v"(lds), "v"(ga) : "memory");
    }
}

__device__ inline void wait_async_all()
{
    asm volatile("s_wait_asynccnt 0x0" ::: "memory");
}

// --------------------------- BN fold + f16 convert -------------------------
__global__ void fold_weight_kernel(const float* __restrict__ W,
                                   const float* __restrict__ g,
                                   const float* __restrict__ bsrc,
                                   const float* __restrict__ be,
                                   h16* __restrict__ Wdst,
                                   float* __restrict__ bdst,
                                   int rows, int cols, int padK)
{
    int gid = blockIdx.x * blockDim.x + threadIdx.x;
    if (gid >= rows * padK) return;
    int r = gid / padK;
    int c = gid - r * padK;
    float s = g ? g[r] * rsqrtf(1.0f + 1e-5f) : 1.0f;
    float v = (c < cols) ? s * W[(size_t)r * cols + c] : 0.0f;
    Wdst[gid] = (h16)v;
    if (c == 0 && bdst) {
        float b0 = bsrc ? bsrc[r] : 0.0f;
        float bev = be ? be[r] : 0.0f;
        bdst[r] = s * b0 + bev;
    }
}

// ------------------------------- three-NN ----------------------------------
__global__ __launch_bounds__(256) void three_nn_kernel(
    const float* __restrict__ up_xyz, const float* __restrict__ xyz,
    int* __restrict__ idx, float* __restrict__ wts)
{
    __shared__ float sx[MPTS], sy[MPTS], sz[MPTS];
    int b = blockIdx.y;
    const float* xb = xyz + (size_t)b * MPTS * 3;
    for (int s = threadIdx.x; s < MPTS; s += 256) {
        sx[s] = xb[s * 3 + 0];
        sy[s] = xb[s * 3 + 1];
        sz[s] = xb[s * 3 + 2];
    }
    __syncthreads();
    int n = blockIdx.x * 256 + threadIdx.x;
    int g = b * NPTS + n;
    float px = up_xyz[g * 3 + 0], py = up_xyz[g * 3 + 1], pz = up_xyz[g * 3 + 2];
    float d0 = 1e30f, d1 = 1e30f, d2 = 1e30f;
    int i0 = 0, i1 = 0, i2 = 0;
    for (int m = 0; m < MPTS; ++m) {
        float dx = px - sx[m], dy = py - sy[m], dz = pz - sz[m];
        float d = dx * dx + dy * dy + dz * dz;
        if (d < d0)      { d2 = d1; i2 = i1; d1 = d0; i1 = i0; d0 = d; i0 = m; }
        else if (d < d1) { d2 = d1; i2 = i1; d1 = d;  i1 = m; }
        else if (d < d2) { d2 = d;  i2 = m; }
    }
    float r0 = 1.0f / fmaxf(d0, 1e-10f);
    float r1 = 1.0f / fmaxf(d1, 1e-10f);
    float r2 = 1.0f / fmaxf(d2, 1e-10f);
    float rs = 1.0f / (r0 + r1 + r2);
    idx[g * 3 + 0] = i0; idx[g * 3 + 1] = i1; idx[g * 3 + 2] = i2;
    wts[g * 3 + 0] = r0 * rs; wts[g * 3 + 1] = r1 * rs; wts[g * 3 + 2] = r2 * rs;
}

// ------------------ gather + interpolate -> X1 (BN x CIN, f16) -------------
__global__ __launch_bounds__(256) void interp_kernel(
    const float* __restrict__ feat, const int* __restrict__ idx,
    const float* __restrict__ wts, h16* __restrict__ X1)
{
    int g = blockIdx.x * 256 + threadIdx.x;       // point column = b*NPTS + n
    int b = g >> 11;
    const float* fb = feat + (size_t)b * CIN * MPTS;
    int i0 = idx[g * 3 + 0], i1 = idx[g * 3 + 1], i2 = idx[g * 3 + 2];
    float w0 = wts[g * 3 + 0], w1 = wts[g * 3 + 1], w2 = wts[g * 3 + 2];
    h16* dst = X1 + (size_t)g * CIN;              // contiguous channel row
    for (int c = 0; c < CIN; ++c) {
        const float* fc = fb + (size_t)c * MPTS;
        float v = w0 * fc[i0] + w1 * fc[i1] + w2 * fc[i2];
        dst[c] = (h16)v;
    }
}

// -------------- build v_in (BN x KFU, f16): [pos(3) ++ global(512) ++ 0] ---
__global__ void build_vin_kernel(const float* __restrict__ up_xyz,
                                 const float* __restrict__ gf,
                                 h16* __restrict__ vin)
{
    int g = blockIdx.x * blockDim.x + threadIdx.x;
    if (g >= KFU * BN) return;
    int col = g / KFU;                      // point
    int r = g - col * KFU;                  // channel
    int b = col >> 11, n = col & (NPTS - 1);
    float v = 0.0f;
    if (r < 3)           v = up_xyz[((size_t)b * NPTS + n) * 3 + r];
    else if (r < 3 + CG) v = gf[(size_t)b * CG + (r - 3)];
    vin[g] = (h16)v;
}

// ------------------- WMMA fragment loaders (CDNA5 layouts) -----------------
// A (16x32 f16, MxK): lane m = lane&15, half = lane>>4
//   vgpr v, elem e: k = (v>=4 ? 16 : 0) + (v&3)*2 + e + half*8
__device__ inline v16h load_frag_a(const h16* __restrict__ base, int lane, int rowStride)
{
    int m = lane & 15, hf = lane >> 4;
    v16h r;
#pragma unroll
    for (int v = 0; v < 8; ++v) {
        int k = ((v & 4) << 2) + ((v & 3) << 1) + (hf << 3);
        r[2 * v + 0] = base[m * rowStride + k + 0];
        r[2 * v + 1] = base[m * rowStride + k + 1];
    }
    return r;
}

// B (32x16 f16, KxN), staged as [n][k] row-major: lane n = lane&15
//   vgpr v, elem e: k = (v>=4 ? 16 : 0) + (v&3)*2 + e + (lane>>4)*8 ; zero-pad beyond kValid
__device__ inline v16h load_frag_b(const h16* __restrict__ base, int lane,
                                   int rowStride, int kValid)
{
    int n = lane & 15, hf = lane >> 4;
    v16h r;
#pragma unroll
    for (int v = 0; v < 8; ++v) {
        int k = ((v & 4) << 2) + ((v & 3) << 1) + (hf << 3);
        r[2 * v + 0] = (k     < kValid) ? base[n * rowStride + k]     : (h16)0;
        r[2 * v + 1] = (k + 1 < kValid) ? base[n * rowStride + k + 1] : (h16)0;
    }
    return r;
}

// ---------------------- generic WMMA GEMM + epilogue -----------------------
// out = act( A(MxK f16, row-major) @ X(Ncols x K f16, row-major)^T + bias )
// i.e. out[col][m] = sum_k A[m][k] * X[col][k]  (X rows are points).
// 64x64 block tile, 4 waves, K sliced by 32 with double-buffered async LDS.
__global__ __launch_bounds__(128) void wmma_gemm_kernel(
    const h16* __restrict__ A, const h16* __restrict__ X,
    const float* __restrict__ bias, void* __restrict__ outp,
    int M, int K, int Ncols, int flags, int Nper)
{
    __shared__ __align__(16) h16 Wt[2][64 * 32];
    __shared__ __align__(16) h16 Xt[2][64 * 32];
    int wave = threadIdx.x >> 5, lane = threadIdx.x & 31;
    int rowBase = blockIdx.y * 64, colBase = blockIdx.x * 64;
    v8f acc[4] = {};

    const h16* Abase = A + (size_t)rowBase * K;
    const h16* Xbase = X + (size_t)colBase * K;
    async_copy_tile64x32(Abase, K, Wt[0]);
    async_copy_tile64x32(Xbase, K, Xt[0]);

    int nIter = K >> 5;
    for (int it = 0; it < nIter; ++it) {
        wait_async_all();
        __syncthreads();
        int cur = it & 1;
        if (it + 1 < nIter) {
            int kb = (it + 1) << 5;
            async_copy_tile64x32(Abase + kb, K, Wt[cur ^ 1]);
            async_copy_tile64x32(Xbase + kb, K, Xt[cur ^ 1]);
        }
        // hoist all fragment loads so the 4 WMMAs issue back-to-back
        v16h af = load_frag_a(&Wt[cur][wave * 16 * 32], lane, 32);
        v16h bf[4];
#pragma unroll
        for (int j = 0; j < 4; ++j)
            bf[j] = load_frag_b(&Xt[cur][j * 16 * 32], lane, 32, 32);
#pragma unroll
        for (int j = 0; j < 4; ++j)
            acc[j] = __builtin_amdgcn_wmma_f32_16x16x32_f16(
                false, af, false, bf[j], (short)0, acc[j], false, false);
    }

    // epilogue: D layout m = r + 8*(lane>>4), n(col) = lane&15
    int mSub = (lane >> 4) << 3;
    int nSub = lane & 15;
#pragma unroll
    for (int j = 0; j < 4; ++j) {
        int col = colBase + j * 16 + nSub;
        int m0 = rowBase + wave * 16 + mSub;
        if (flags & FLAG_F16) {
            h16 ov[8];
#pragma unroll
            for (int r = 0; r < 8; ++r) {
                float v = acc[j][r] + (bias ? bias[m0 + r] : 0.0f);
                if (flags & FLAG_RELU) v = fmaxf(v, 0.0f);
                ov[r] = (h16)v;
            }
            h16* dst = (h16*)outp + (size_t)col * M + m0;
            *(uint4*)dst = *(const uint4*)ov;        // 8 contiguous channels
        } else {
#pragma unroll
            for (int r = 0; r < 8; ++r) {
                int m = m0 + r;
                float v = acc[j][r] + (bias ? bias[m] : 0.0f);
                if (flags & FLAG_RELU) v = fmaxf(v, 0.0f);
                if (flags & FLAG_BATCH) {
                    int bb = col / Nper, nn = col - bb * Nper;
                    ((float*)outp)[((size_t)bb * M + m) * Nper + nn] = v;
                } else {
                    ((float*)outp)[(size_t)col * M + m] = v;
                }
            }
        }
    }
}

// ----------------------- flash attention (WMMA + online softmax) -----------
// Q/K/V/O are (BN x COUT) row-major; head h uses channels [h*16, h*16+16).
// grid: (NPTS/64, BATCH*NHEADS), 4 waves; wave owns 16 query rows.
__global__ __launch_bounds__(128) void flash_attn_kernel(
    const h16* __restrict__ Q, const h16* __restrict__ Kmat,
    const h16* __restrict__ Vmat, h16* __restrict__ Omat)
{
    __shared__ __align__(16) h16 Ks[32][16];      // [key][d] (async DMA)
    __shared__ __align__(16) h16 Vs[16][32];      // [d][key] (scalar transpose)
    __shared__ __align__(16) h16 Ps[4][16][32];   // per-wave P tile [m][k]
    int wave = threadIdx.x >> 5, lane = threadIdx.x & 31;
    int b = blockIdx.y >> 3, h = blockIdx.y & 7;
    int qcol = b * NPTS + blockIdx.x * 64 + wave * 16;
    int ln = lane & 15, hf = lane >> 4;

    // Q A-fragment, scale (D^-0.5 = 0.25) folded in; K-dim padded 16 -> 32
    const h16* qrow = Q + (size_t)(qcol + ln) * COUT + h * DH;
    v16h qf;
#pragma unroll
    for (int v = 0; v < 8; ++v) {
#pragma unroll
        for (int e = 0; e < 2; ++e) {
            int k = ((v & 4) << 2) + ((v & 3) << 1) + e + (hf << 3);
            float qv = (v < 4) ? 0.25f * (float)qrow[k] : 0.0f;
            qf[2 * v + e] = (h16)qv;
        }
    }

    float rmax[8], rsum[8];
    v8f oacc = {};
#pragma unroll
    for (int r = 0; r < 8; ++r) { rmax[r] = -1e30f; rsum[r] = 0.0f; }

    int bcol = b * NPTS;
    for (int kb = 0; kb < NPTS; kb += 32) {
        __syncthreads();
        // K tile: straight async DMA (32 keys x 16 halves = 64 x 16B chunks)
        if (threadIdx.x < 64) {
            int key = threadIdx.x >> 1, seg = threadIdx.x & 1;
            const h16* g = Kmat + (size_t)(bcol + kb + key) * COUT + h * DH + seg * 8;
            unsigned lds = (unsigned)(uintptr_t)(&Ks[key][seg * 8]);
            unsigned long long ga = (unsigned long long)(uintptr_t)g;
            asm volatile("global_load_async_to_lds_b128 %0, %1, off"
                         :: "v"(lds), "v"(ga) : "memory");
        }
        // V tile: transposed staging [d][key]
        for (int s = threadIdx.x; s < 512; s += 128) {
            int key = s & 31, d = s >> 5;
            Vs[d][key] = Vmat[(size_t)(bcol + kb + key) * COUT + h * DH + d];
        }
        wait_async_all();
        __syncthreads();

        // S = scaled Q @ K^T for 32 keys (two 16-key sub-tiles); also pre-load
        // the V fragment so the later PV WMMA only waits on the P round-trip.
        v16h bf0 = load_frag_b(&Ks[0][0],  lane, 16, 16);
        v16h bf1 = load_frag_b(&Ks[16][0], lane, 16, 16);
        v16h vf  = load_frag_b(&Vs[0][0],  lane, 32, 32);
        v8f zero = {};
        v8f s0 = __builtin_amdgcn_wmma_f32_16x16x32_f16(false, qf, false, bf0, (short)0, zero, false, false);
        v8f s1 = __builtin_amdgcn_wmma_f32_16x16x32_f16(false, qf, false, bf1, (short)0, zero, false, false);

        // online softmax over keys: row m lives in one 16-lane half
        float cmax[8];
#pragma unroll
        for (int r = 0; r < 8; ++r) cmax[r] = fmaxf(s0[r], s1[r]);
#pragma unroll
        for (int d = 1; d < 16; d <<= 1)
#pragma unroll
            for (int r = 0; r < 8; ++r)
                cmax[r] = fmaxf(cmax[r], __shfl_xor(cmax[r], d, 16));

        float p0[8], p1[8], csum[8], alpha[8];
#pragma unroll
        for (int r = 0; r < 8; ++r) {
            float nm = fmaxf(rmax[r], cmax[r]);
            alpha[r] = __expf(rmax[r] - nm);
            rmax[r]  = nm;
            p0[r] = __expf(s0[r] - nm);
            p1[r] = __expf(s1[r] - nm);
            csum[r] = p0[r] + p1[r];
        }
#pragma unroll
        for (int d = 1; d < 16; d <<= 1)
#pragma unroll
            for (int r = 0; r < 8; ++r)
                csum[r] += __shfl_xor(csum[r], d, 16);
#pragma unroll
        for (int r = 0; r < 8; ++r) {
            rsum[r] = rsum[r] * alpha[r] + csum[r];
            oacc[r] = oacc[r] * alpha[r];
        }

        // D-layout P -> per-wave LDS (row-major 16x32) -> A-layout fragment
#pragma unroll
        for (int r = 0; r < 8; ++r) {
            int mm = r + (hf << 3);
            Ps[wave][mm][ln]      = (h16)p0[r];
            Ps[wave][mm][ln + 16] = (h16)p1[r];
        }
        v16h pf = load_frag_a(&Ps[wave][0][0], lane, 32);
        oacc = __builtin_amdgcn_wmma_f32_16x16x32_f16(false, pf, false, vf, (short)0, oacc, false, false);
    }

    // normalize + store: O[(qcol+m)][h*16 + d], d = lane&15, m = r + 8*hf
#pragma unroll
    for (int r = 0; r < 8; ++r) {
        h16 ov = (h16)(oacc[r] / rsum[r]);
        Omat[(size_t)(qcol + (hf << 3) + r) * COUT + h * DH + ln] = ov;
    }
}

// ------------------------------ residual add -------------------------------
__global__ void add_h16_kernel(const h16* __restrict__ a, const h16* __restrict__ b,
                               h16* __restrict__ c, int n)
{
    int g = blockIdx.x * blockDim.x + threadIdx.x;
    if (g < n) c[g] = a[g] + b[g];
}

// ------------------------------- launcher ----------------------------------
extern "C" void kernel_launch(void* const* d_in, const int* in_sizes, int n_in,
                              void* d_out, int out_size, void* d_ws, size_t ws_size,
                              hipStream_t stream)
{
    (void)in_sizes; (void)n_in; (void)out_size; (void)ws_size;
    const float* up_xyz = (const float*)d_in[0];
    const float* xyz    = (const float*)d_in[1];
    const float* feat   = (const float*)d_in[2];
    const float* gfeat  = (const float*)d_in[3];
    const float* fp1_W  = (const float*)d_in[4];
    const float* fp1_b  = (const float*)d_in[5];
    const float* fp1_g  = (const float*)d_in[6];
    const float* fp1_be = (const float*)d_in[7];
    const float* fp2_W  = (const float*)d_in[8];
    const float* fp2_b  = (const float*)d_in[9];
    const float* fp2_g  = (const float*)d_in[10];
    const float* fp2_be = (const float*)d_in[11];
    const float* qm_W   = (const float*)d_in[12];
    const float* qm_b   = (const float*)d_in[13];
    const float* qm_g   = (const float*)d_in[14];
    const float* qm_be  = (const float*)d_in[15];
    const float* fu_W   = (const float*)d_in[16];
    const float* fu_b   = (const float*)d_in[17];
    const float* fu_g   = (const float*)d_in[18];
    const float* fu_be  = (const float*)d_in[19];
    const float* Wq     = (const float*)d_in[20];
    const float* Wk     = (const float*)d_in[21];
    const float* Wv     = (const float*)d_in[22];
    const float* Wp     = (const float*)d_in[23];
    const float* bp     = (const float*)d_in[24];
    const float* om_W   = (const float*)d_in[25];
    const float* om_b   = (const float*)d_in[26];
    const float* om_g   = (const float*)d_in[27];
    const float* om_be  = (const float*)d_in[28];

    char* ws = (char*)d_ws;
    size_t off = 0;
    auto alloc = [&](size_t bytes) -> char* {
        char* p = ws + off;
        off = (off + bytes + 255) & ~(size_t)255;
        return p;
    };

    h16* fp1f = (h16*)alloc((size_t)CIN * CIN * 2);
    h16* fp2f = (h16*)alloc((size_t)CFP * CIN * 2);
    h16* qmf  = (h16*)alloc((size_t)COUT * CFP * 2);
    h16* fuf  = (h16*)alloc((size_t)COUT * KFU * 2);
    h16* wqf  = (h16*)alloc((size_t)COUT * COUT * 2);
    h16* wkf  = (h16*)alloc((size_t)COUT * COUT * 2);
    h16* wvf  = (h16*)alloc((size_t)COUT * COUT * 2);
    h16* wpf  = (h16*)alloc((size_t)COUT * COUT * 2);
    h16* omf  = (h16*)alloc((size_t)COUT * COUT * 2);
    float* bias = (float*)alloc(896 * sizeof(float));
    // bias offsets: b1:0(256) b2:256 bqm:384 bfu:512 bp:640 bom:768
    int*   idxb = (int*)alloc((size_t)BN * 3 * sizeof(int));
    float* wtsb = (float*)alloc((size_t)BN * 3 * sizeof(float));
    h16* X1region  = (h16*)alloc((size_t)CIN * BN * 2);   // 8 MB
    h16* VinRegion = (h16*)alloc((size_t)KFU * BN * 2);   // 17 MB
    h16* vfeat     = (h16*)alloc((size_t)COUT * BN * 2);  // 4 MB

    // region aliasing (lifetimes disjoint; all activations are BN x K rows)
    h16* X1     = X1region;                         // BN x CIN
    h16* qfeatB = X1region;                         // BN x COUT (X1 dead)
    h16* attnB  = X1region + (size_t)COUT * BN;     // BN x COUT
    h16* vin    = VinRegion;                        // BN x KFU
    h16* H1     = VinRegion;                        // BN x CIN  (v_in dead)
    h16* X2     = VinRegion + (size_t)CIN * BN;     // BN x CFP
    h16* Qm     = VinRegion;                        // BN x COUT (H1 dead)
    h16* Km     = VinRegion + (size_t)COUT * BN;
    h16* Vm     = VinRegion + (size_t)2 * COUT * BN;
    h16* transB = VinRegion + (size_t)3 * COUT * BN;
    h16* sumB   = VinRegion;                        // (Qm dead)

    auto g1 = [](int total) { return dim3((total + 255) / 256); };
    dim3 b256(256), b128(128);

    // 1) fold BN into weights, convert to f16 (pad K where needed)
    fold_weight_kernel<<<g1(CIN * CIN), b256, 0, stream>>>(fp1_W, fp1_g, fp1_b, fp1_be, fp1f, bias + 0,   CIN,  CIN,  CIN);
    fold_weight_kernel<<<g1(CFP * CIN), b256, 0, stream>>>(fp2_W, fp2_g, fp2_b, fp2_be, fp2f, bias + 256, CFP,  CIN,  CIN);
    fold_weight_kernel<<<g1(COUT * CFP), b256, 0, stream>>>(qm_W, qm_g, qm_b, qm_be,   qmf,  bias + 384, COUT, CFP,  CFP);
    fold_weight_kernel<<<g1(COUT * KFU), b256, 0, stream>>>(fu_W, fu_g, fu_b, fu_be,   fuf,  bias + 512, COUT, 3 + CG, KFU);
    fold_weight_kernel<<<g1(COUT * COUT), b256, 0, stream>>>(Wq, nullptr, nullptr, nullptr, wqf, nullptr, COUT, COUT, COUT);
    fold_weight_kernel<<<g1(COUT * COUT), b256, 0, stream>>>(Wk, nullptr, nullptr, nullptr, wkf, nullptr, COUT, COUT, COUT);
    fold_weight_kernel<<<g1(COUT * COUT), b256, 0, stream>>>(Wv, nullptr, nullptr, nullptr, wvf, nullptr, COUT, COUT, COUT);
    fold_weight_kernel<<<g1(COUT * COUT), b256, 0, stream>>>(Wp, nullptr, bp, nullptr,  wpf, bias + 640, COUT, COUT, COUT);
    fold_weight_kernel<<<g1(COUT * COUT), b256, 0, stream>>>(om_W, om_g, om_b, om_be,  omf,  bias + 768, COUT, COUT, COUT);

    // 2) fuse branch first (so v_in's region can be recycled)
    build_vin_kernel<<<g1(KFU * BN), b256, 0, stream>>>(up_xyz, gfeat, vin);
    wmma_gemm_kernel<<<dim3(BN / 64, COUT / 64), b128, 0, stream>>>(
        fuf, vin, bias + 512, vfeat, COUT, KFU, BN, FLAG_RELU | FLAG_F16, NPTS);

    // 3) FP interpolation + mlp chain
    three_nn_kernel<<<dim3(NPTS / 256, BATCH), b256, 0, stream>>>(up_xyz, xyz, idxb, wtsb);
    interp_kernel<<<g1(BN), b256, 0, stream>>>(feat, idxb, wtsb, X1);
    wmma_gemm_kernel<<<dim3(BN / 64, CIN / 64), b128, 0, stream>>>(
        fp1f, X1, bias + 0, H1, CIN, CIN, BN, FLAG_RELU | FLAG_F16, NPTS);
    wmma_gemm_kernel<<<dim3(BN / 64, CFP / 64), b128, 0, stream>>>(
        fp2f, H1, bias + 256, X2, CFP, CIN, BN, FLAG_RELU | FLAG_F16, NPTS);
    wmma_gemm_kernel<<<dim3(BN / 64, COUT / 64), b128, 0, stream>>>(
        qmf, X2, bias + 384, qfeatB, COUT, CFP, BN, FLAG_RELU | FLAG_F16, NPTS);

    // 4) attention linears
    wmma_gemm_kernel<<<dim3(BN / 64, COUT / 64), b128, 0, stream>>>(
        wqf, qfeatB, nullptr, Qm, COUT, COUT, BN, FLAG_F16, NPTS);
    wmma_gemm_kernel<<<dim3(BN / 64, COUT / 64), b128, 0, stream>>>(
        wkf, vfeat, nullptr, Km, COUT, COUT, BN, FLAG_F16, NPTS);
    wmma_gemm_kernel<<<dim3(BN / 64, COUT / 64), b128, 0, stream>>>(
        wvf, vfeat, nullptr, Vm, COUT, COUT, BN, FLAG_F16, NPTS);

    // 5) flash attention
    flash_attn_kernel<<<dim3(NPTS / 64, BATCH * NHEADS), b128, 0, stream>>>(Qm, Km, Vm, attnB);

    // 6) projection + residual + output mlp (batched f32 layout (B,C,N))
    wmma_gemm_kernel<<<dim3(BN / 64, COUT / 64), b128, 0, stream>>>(
        wpf, attnB, bias + 640, transB, COUT, COUT, BN, FLAG_F16, NPTS);
    add_h16_kernel<<<g1(COUT * BN), b256, 0, stream>>>(qfeatB, transB, sumB, COUT * BN);
    wmma_gemm_kernel<<<dim3(BN / 64, COUT / 64), b128, 0, stream>>>(
        omf, sumB, bias + 768, d_out, COUT, COUT, BN, FLAG_RELU | FLAG_BATCH, NPTS);
}
